// Seq2SeqTransformer_35313221108095
// MI455X (gfx1250) — compile-verified, matrix operand order checked by reading, not run on hardware
//
#include <hip/hip_runtime.h>
#include <hip/hip_bf16.h>
#include <math.h>

// ---------------------------------------------------------------------------
// MI455X (gfx1250) seq2seq transformer greedy decode.
//   - fp16 weights staged once per launch into d_ws (~120MB, fits in 192MB L2)
//   - all GEMMs via v_wmma_f32_16x16x32_f16; MT=2 M-tiles per wave to halve
//     weight re-reads from L2 (decode loop re-streams weights 31x)
//   - fused per-(batch,head) attention with TDM (tensor_load_to_lds) staging
//     of Q/K/V tiles into LDS + s_wait_tensorcnt, WMMA for QK^T and P*V
//   - cross-attn K/V precomputed once (encoder memory is loop-invariant)
// ---------------------------------------------------------------------------

typedef _Float16 h16v __attribute__((ext_vector_type(16)));
typedef _Float16 h8v  __attribute__((ext_vector_type(8)));
typedef float    f8v  __attribute__((ext_vector_type(8)));
typedef unsigned int u32x4 __attribute__((ext_vector_type(4)));
typedef int          i32x4 __attribute__((ext_vector_type(4)));
typedef int          i32x8 __attribute__((ext_vector_type(8)));

#define E_    512
#define NH_   8
#define DH_   64
#define FF_   2048
#define T_    32
#define S_    32
#define B_    8
#define V_    32000
#define ROWS_ 256   // T_*B_ == S_*B_

#if defined(__has_builtin)
#  if __has_builtin(__builtin_amdgcn_tensor_load_to_lds) && \
      __has_builtin(__builtin_amdgcn_s_wait_tensorcnt)
#    define USE_TDM 1
#  endif
#endif
#ifndef USE_TDM
#  define USE_TDM 0
#endif

// ---- WMMA fragment loaders -------------------------------------------------
// 16-bit A-fragment layout (ISA 7.12.2): lanes 0-15 hold row m, K in
// {kb..kb+7, kb+16..kb+23}; lanes 16-31 the complementary K bytes. Both X and
// W (stored [N,K]) are K-contiguous, so a fragment is two 16B loads per lane.
__device__ __forceinline__ h16v frag_ld_k(const _Float16* base, int ld, int row0, int k0) {
  const int lane = threadIdx.x & 31;
  const int r    = lane & 15;
  const int kb   = (lane >> 4) << 3;
  const _Float16* p = base + (size_t)(row0 + r) * ld + k0 + kb;
  h8v lo = *(const h8v*)p;
  h8v hi = *(const h8v*)(p + 16);
  return __builtin_shufflevector(lo, hi, 0,1,2,3,4,5,6,7,8,9,10,11,12,13,14,15);
}

// B-fragment when B is stored row-major [K,N] (V in attention):
// element (k,n) = base[k*ld + col0 + n]; per-lane gather along k (LDS).
__device__ __forceinline__ h16v frag_ld_kn(const _Float16* base, int ld, int col0, int k0) {
  const int lane = threadIdx.x & 31;
  const int n    = lane & 15;
  const int kb   = (lane >> 4) << 3;
  h16v v;
#pragma unroll
  for (int e = 0; e < 16; ++e) {
    int k = k0 + kb + (e & 7) + ((e >> 3) << 4);
    v[e] = base[(size_t)k * ld + col0 + n];
  }
  return v;
}

// ---- f32 -> f16 weight staging --------------------------------------------
__global__ void k_cvt(const float* __restrict__ s, _Float16* __restrict__ d, int n) {
  for (int i = blockIdx.x * blockDim.x + threadIdx.x; i < n; i += gridDim.x * blockDim.x)
    d[i] = (_Float16)s[i];
}

// ---- GEMM: Y[M,N] = act(X[M,K] * W[N,K]^T + bias) --------------------------
// 4 waves/block; each wave owns MT vertically-stacked 16x16 tiles so one
// B-fragment feeds MT WMMAs (halves weight traffic from L2 for MT=2).
template <int MT>
__global__ __launch_bounds__(128) void k_gemm(const _Float16* __restrict__ A,
    const _Float16* __restrict__ W, const float* __restrict__ bias,
    float* __restrict__ o32, _Float16* __restrict__ o16,
    int M, int N, int K, int relu)
{
  const int wave = threadIdx.x >> 5;
  const int lane = threadIdx.x & 31;
  const int nb  = (blockIdx.x * 4 + wave) * 16;
  const int mb0 = blockIdx.y * (16 * MT);
  if (nb >= N) return;
  f8v acc[MT] = {};
  for (int k = 0; k < K; k += 32) {
    h16v b = frag_ld_k(W, K, nb, k);
    if (k + 32 < K) {   // keep the next K-slab warm in near caches (RT, WGP)
      __builtin_prefetch(W + (size_t)(nb + (lane & 15)) * K + k + 32, 0, 3);
      __builtin_prefetch(A + (size_t)(mb0 + (lane & 15)) * K + k + 32, 0, 3);
    }
#pragma unroll
    for (int t = 0; t < MT; ++t) {
      h16v a = frag_ld_k(A, K, mb0 + 16 * t, k);
      acc[t] = __builtin_amdgcn_wmma_f32_16x16x32_f16(false, a, false, b, (short)0,
                                                      acc[t], false, false);
    }
  }
  const int col = nb + (lane & 15);
  const int rh  = lane >> 4;
  const float bv = bias ? bias[col] : 0.0f;
#pragma unroll
  for (int t = 0; t < MT; ++t) {
#pragma unroll
    for (int r = 0; r < 8; ++r) {
      int row = mb0 + 16 * t + r + 8 * rh;
      if (row >= M) continue;
      float v = acc[t][r] + bv;
      if (relu) v = fmaxf(v, 0.0f);
      size_t idx = (size_t)row * N + col;
      if (o32) o32[idx] = v;
      if (o16) o16[idx] = (_Float16)v;
    }
  }
}

#if USE_TDM
// ---- Tensor Data Mover: 2D tile (32 rows x 128B, row stride 8192B) -> LDS --
// D# per ISA 8.3/8.4: group0 = {count|flags, lds_addr, global_addr, type=2};
// group1 packs data_size=2B, tensor/tile dims, dim0 stride (data_size units).
__device__ __forceinline__ void tdm_load_tile(unsigned lds_off, const void* gptr) {
  unsigned long long ga = (unsigned long long)(uintptr_t)gptr;
  u32x4 g0;
  g0[0] = 1u;                                            // count=1 (user D#)
  g0[1] = lds_off;                                       // lds_addr (bytes)
  g0[2] = (unsigned)(ga & 0xffffffffu);                  // global_addr[31:0]
  g0[3] = (unsigned)((ga >> 32) & 0x01ffffffu) | (2u << 30); // [56:32] | type=2
  i32x8 g1 = {0, 0, 0, 0, 0, 0, 0, 0};
  g1[0] = (1 << 16);           // workgroup_mask=0, data_size=1 (2 bytes)
  g1[1] = (DH_ << 16);         // tensor_dim0 = 64 elements (row length)
  g1[2] = (32 << 16);          // tensor_dim0 hi=0 | tensor_dim1 = 32 rows
  g1[3] = (DH_ << 16);         // tensor_dim1 hi=0 | tile_dim0 = 64
  g1[4] = 32;                  // tile_dim1 = 32, tile_dim2 = 0
  g1[5] = (B_ * E_);           // tensor_dim0_stride = 4096 elements
  g1[6] = 0;                   // stride hi, tensor_dim1_stride lo
  g1[7] = 0;
  i32x4 z4 = {0, 0, 0, 0};
#if __clang_major__ >= 23
  i32x8 z8 = {0, 0, 0, 0, 0, 0, 0, 0};
  __builtin_amdgcn_tensor_load_to_lds(g0, g1, z4, z4, z8, 0);
#else
  __builtin_amdgcn_tensor_load_to_lds(g0, g1, z4, z4, 0);
#endif
}
#endif

// ---- Fused attention per (b,h): S=QK^T/8 (+causal), softmax, O=P*V ---------
// 2 waves; wave w owns M-rows [16w,16w+16). Dynamic LDS so TDM lds offsets
// are exact: Q@0, K@4096, V@8192, S@12288, P@16512 (total 18560 bytes).
__global__ __launch_bounds__(64) void k_attn(const _Float16* __restrict__ Q,
    const _Float16* __restrict__ Km, const _Float16* __restrict__ Vm,
    _Float16* __restrict__ O, int causal)
{
  extern __shared__ char smem[];
  _Float16* Qs = (_Float16*)(smem);
  _Float16* Ks = (_Float16*)(smem + 4096);
  _Float16* Vs = (_Float16*)(smem + 8192);
  float*    Ss = (float*)   (smem + 12288);   // 32 x 33 f32
  _Float16* Ps = (_Float16*)(smem + 16512);   // 32 x 32 f16
  const int h   = blockIdx.x & (NH_ - 1);
  const int b   = blockIdx.x >> 3;
  const int tid = threadIdx.x;
  const size_t gbase = (size_t)b * E_ + (size_t)h * DH_;   // row l=0
#if USE_TDM
  if (tid < 32) {   // one wave issues the three async tile DMAs
    tdm_load_tile(0u,    (const void*)(Q  + gbase));
    tdm_load_tile(4096u, (const void*)(Km + gbase));
    tdm_load_tile(8192u, (const void*)(Vm + gbase));
    __builtin_amdgcn_s_wait_tensorcnt((short)0);
  }
  __syncthreads();
#else
  for (int i = tid; i < 32 * DH_; i += 64) {
    int l = i >> 6, d = i & 63;
    size_t g = gbase + (size_t)l * (B_ * E_) + d;
    Qs[i] = Q[g]; Ks[i] = Km[g]; Vs[i] = Vm[g];
  }
  __syncthreads();
#endif
  const int wave = tid >> 5, lane = tid & 31;
  const int mb = wave * 16;
  const int cn = lane & 15, rh = lane >> 4;
  // scores S = Q K^T * (1/sqrt(DH))
  for (int nt = 0; nt < 2; ++nt) {
    f8v acc = {};
    for (int k = 0; k < DH_; k += 32) {
      h16v a  = frag_ld_k(Qs, DH_, mb, k);
      h16v bb = frag_ld_k(Ks, DH_, nt * 16, k);   // B[k][n] = K[n][k]
      acc = __builtin_amdgcn_wmma_f32_16x16x32_f16(false, a, false, bb, (short)0, acc,
                                                   false, false);
    }
#pragma unroll
    for (int r = 0; r < 8; ++r)
      Ss[(mb + r + 8 * rh) * 33 + nt * 16 + cn] = acc[r] * 0.125f;
  }
  __syncthreads();
  // row softmax (causal: cols <= row)
  if (tid < 32) {
    const int row = tid;
    const int lim = causal ? (row + 1) : 32;
    float mx = -3.4e38f;
    for (int c2 = 0; c2 < lim; ++c2) mx = fmaxf(mx, Ss[row * 33 + c2]);
    float sum = 0.0f;
    for (int c2 = 0; c2 < lim; ++c2) {
      float e = __expf(Ss[row * 33 + c2] - mx);
      Ss[row * 33 + c2] = e; sum += e;
    }
    float inv = 1.0f / sum;
    for (int c2 = 0; c2 < 32; ++c2)
      Ps[row * 32 + c2] = (_Float16)(c2 < lim ? Ss[row * 33 + c2] * inv : 0.0f);
  }
  __syncthreads();
  // O = P[32,32] * V[32,64]
  for (int nt = 0; nt < 4; ++nt) {
    f8v acc = {};
    h16v a  = frag_ld_k(Ps, 32, mb, 0);
    h16v bb = frag_ld_kn(Vs, DH_, nt * 16, 0);    // B[k][n] = V[k][n]
    acc = __builtin_amdgcn_wmma_f32_16x16x32_f16(false, a, false, bb, (short)0, acc,
                                                 false, false);
#pragma unroll
    for (int r = 0; r < 8; ++r) {
      int l = mb + r + 8 * rh;
      int d = nt * 16 + cn;
      O[(size_t)(l * B_ + b) * E_ + h * DH_ + d] = (_Float16)acc[r];
    }
  }
}

// ---- residual + LayerNorm (y may be null -> pure LN of xres) ---------------
__global__ __launch_bounds__(256) void k_res_ln(const float* __restrict__ y,
    const _Float16* __restrict__ xr, const float* __restrict__ w,
    const float* __restrict__ bb, _Float16* __restrict__ out)
{
  const int row = blockIdx.x;
  const int t   = threadIdx.x;
  const int i0 = t * 2, i1 = t * 2 + 1;
  const size_t base = (size_t)row * E_;
  float v0 = (float)xr[base + i0];
  float v1 = (float)xr[base + i1];
  if (y) { v0 += y[base + i0]; v1 += y[base + i1]; }
  float s = v0 + v1, sq = v0 * v0 + v1 * v1;
#pragma unroll
  for (int m = 16; m >= 1; m >>= 1) {
    s  += __shfl_xor(s,  m, 32);
    sq += __shfl_xor(sq, m, 32);
  }
  __shared__ float ss[8], ssq[8];
  if ((t & 31) == 0) { ss[t >> 5] = s; ssq[t >> 5] = sq; }
  __syncthreads();
  float S = 0.0f, SQ = 0.0f;
#pragma unroll
  for (int k = 0; k < 8; ++k) { S += ss[k]; SQ += ssq[k]; }
  const float mean = S * (1.0f / E_);
  const float var  = SQ * (1.0f / E_) - mean * mean;
  const float inv  = rsqrtf(var + 1e-5f);
  out[base + i0] = (_Float16)((v0 - mean) * inv * w[i0] + bb[i0]);
  out[base + i1] = (_Float16)((v1 - mean) * inv * w[i1] + bb[i1]);
}

// ---- embeddings ------------------------------------------------------------
__global__ __launch_bounds__(256) void k_embed_enc(const int* __restrict__ tok,
    const float* __restrict__ emb, _Float16* __restrict__ x)
{
  const int sb = blockIdx.x;        // s*B + b
  const int s = sb / B_, b = sb % B_;
  const int tk = tok[b * S_ + s];   // input is [B,S]
  for (int e = threadIdx.x; e < E_; e += blockDim.x) {
    int i2 = e >> 1;
    float div = __expf((float)(2 * i2) * (-9.210340371976184f / (float)E_));
    float ang = (float)s * div;
    float pe  = (e & 1) ? __cosf(ang) : __sinf(ang);
    x[(size_t)sb * E_ + e] = (_Float16)(emb[(size_t)tk * E_ + e] + pe);
  }
}

__global__ __launch_bounds__(256) void k_embed_dec(const int* __restrict__ pred,
    const float* __restrict__ emb, _Float16* __restrict__ x)
{
  const int tb = blockIdx.x;        // t*B + b
  const int tk = pred[tb];
  for (int e = threadIdx.x; e < E_; e += blockDim.x)
    x[(size_t)tb * E_ + e] = (_Float16)emb[(size_t)tk * E_ + e];
}

// ---- init (pred=START, tokens row0, masked one-hot dist row0) --------------
__global__ void k_init(int* __restrict__ pred, float* __restrict__ tok,
                       float* __restrict__ dist0)
{
  const int stride = gridDim.x * blockDim.x;
  const int idx = blockIdx.x * blockDim.x + threadIdx.x;
  for (int i = idx; i < T_ * B_; i += stride) pred[i] = 1;
  if (idx < B_) tok[idx] = 1.0f;
  for (int i = idx; i < B_ * V_; i += stride)
    dist0[i] = ((i % V_) == 1) ? 0.0f : -__builtin_inff();
}

// ---- argmax over V per batch row -------------------------------------------
__global__ __launch_bounds__(256) void k_argmax(const float* __restrict__ logits,
    int* __restrict__ pred, float* __restrict__ tok, int i)
{
  const int b = blockIdx.x;
  const float* row = logits + (size_t)b * V_;
  float best = -3.4e38f; int bi = V_;
  for (int v = threadIdx.x; v < V_; v += 256) {
    float f = row[v];
    if (f > best || (f == best && v < bi)) { best = f; bi = v; }
  }
  __shared__ float sv[256]; __shared__ int si[256];
  sv[threadIdx.x] = best; si[threadIdx.x] = bi;
  __syncthreads();
  for (int s = 128; s > 0; s >>= 1) {
    if (threadIdx.x < s) {
      float ov = sv[threadIdx.x + s]; int oi = si[threadIdx.x + s];
      if (ov > sv[threadIdx.x] || (ov == sv[threadIdx.x] && oi < si[threadIdx.x])) {
        sv[threadIdx.x] = ov; si[threadIdx.x] = oi;
      }
    }
    __syncthreads();
  }
  if (threadIdx.x == 0) {
    pred[(i + 1) * B_ + b] = si[0];
    tok[(i + 1) * B_ + b]  = (float)si[0];
  }
}

// ---------------------------------------------------------------------------
extern "C" void kernel_launch(void* const* d_in, const int* in_sizes, int n_in,
                              void* d_out, int out_size, void* d_ws, size_t ws_size,
                              hipStream_t stream)
{
  (void)in_sizes; (void)n_in; (void)out_size; (void)ws_size;
  int c = 0;
  const int*   input_tokens = (const int*)d_in[c++];
  const float* enc_emb      = (const float*)d_in[c++];
  const float* dec_emb      = (const float*)d_in[c++];

  struct AP { const float *wq,*wk,*wv,*wo,*bq,*bk,*bv,*bo; };
  struct EL { AP sa; const float *w1,*b1,*w2,*b2,*ln1w,*ln1b,*ln2w,*ln2b; };
  struct DL { AP sa, ca; const float *w1,*b1,*w2,*b2,*ln1w,*ln1b,*ln2w,*ln2b,*ln3w,*ln3b; };
  auto getf  = [&]() -> const float* { return (const float*)d_in[c++]; };
  auto getAP = [&]() { AP a; a.wq=getf(); a.wk=getf(); a.wv=getf(); a.wo=getf();
                       a.bq=getf(); a.bk=getf(); a.bv=getf(); a.bo=getf(); return a; };
  EL enc[6]; DL dec[6];
  for (int l = 0; l < 6; ++l) {
    enc[l].sa = getAP();
    enc[l].w1 = getf(); enc[l].b1 = getf(); enc[l].w2 = getf(); enc[l].b2 = getf();
    enc[l].ln1w = getf(); enc[l].ln1b = getf(); enc[l].ln2w = getf(); enc[l].ln2b = getf();
  }
  for (int l = 0; l < 6; ++l) {
    dec[l].sa = getAP(); dec[l].ca = getAP();
    dec[l].w1 = getf(); dec[l].b1 = getf(); dec[l].w2 = getf(); dec[l].b2 = getf();
    dec[l].ln1w = getf(); dec[l].ln1b = getf(); dec[l].ln2w = getf(); dec[l].ln2b = getf();
    dec[l].ln3w = getf(); dec[l].ln3b = getf();
  }
  const float* enc_nw = getf(); const float* enc_nb = getf();
  const float* dec_nw = getf(); const float* dec_nb = getf();
  const float* proj_w = getf(); const float* proj_b = getf();

  // bump allocator in workspace
  char* wsb = (char*)d_ws;
  size_t off = 0;
  auto alloc = [&](size_t bytes) -> char* {
    char* p = wsb + off;
    off = (off + bytes + 255) & ~(size_t)255;
    return p;
  };
  auto cvt = [&](const float* src, size_t n) -> _Float16* {
    _Float16* dst = (_Float16*)alloc(n * sizeof(_Float16));
    int blocks = (int)((n + 255) / 256); if (blocks > 2048) blocks = 2048;
    k_cvt<<<blocks, 256, 0, stream>>>(src, dst, (int)n);
    return dst;
  };

  struct AP16 { _Float16 *wq,*wk,*wv,*wo; };
  struct EL16 { AP16 sa; _Float16 *w1,*w2; };
  struct DL16 { AP16 sa, ca; _Float16 *w1,*w2; };
  auto cvtAP = [&](const AP& a) {
    AP16 r;
    r.wq = cvt(a.wq, (size_t)E_ * E_); r.wk = cvt(a.wk, (size_t)E_ * E_);
    r.wv = cvt(a.wv, (size_t)E_ * E_); r.wo = cvt(a.wo, (size_t)E_ * E_);
    return r;
  };
  EL16 enc16[6]; DL16 dec16[6];
  for (int l = 0; l < 6; ++l) {
    enc16[l].sa = cvtAP(enc[l].sa);
    enc16[l].w1 = cvt(enc[l].w1, (size_t)FF_ * E_);
    enc16[l].w2 = cvt(enc[l].w2, (size_t)E_ * FF_);
  }
  for (int l = 0; l < 6; ++l) {
    dec16[l].sa = cvtAP(dec[l].sa);
    dec16[l].ca = cvtAP(dec[l].ca);
    dec16[l].w1 = cvt(dec[l].w1, (size_t)FF_ * E_);
    dec16[l].w2 = cvt(dec[l].w2, (size_t)E_ * FF_);
  }
  _Float16* proj16 = cvt(proj_w, (size_t)V_ * E_);

  // activations
  _Float16* xe  = (_Float16*)alloc((size_t)ROWS_ * E_ * 2);
  _Float16* mem = (_Float16*)alloc((size_t)ROWS_ * E_ * 2);
  _Float16* qb  = (_Float16*)alloc((size_t)ROWS_ * E_ * 2);
  _Float16* kb  = (_Float16*)alloc((size_t)ROWS_ * E_ * 2);
  _Float16* vb  = (_Float16*)alloc((size_t)ROWS_ * E_ * 2);
  _Float16* at  = (_Float16*)alloc((size_t)ROWS_ * E_ * 2);
  _Float16* hb  = (_Float16*)alloc((size_t)ROWS_ * FF_ * 2);
  float*    y32 = (float*)   alloc((size_t)ROWS_ * E_ * 4);
  _Float16* xd  = (_Float16*)alloc((size_t)ROWS_ * E_ * 2);
  _Float16* xn  = (_Float16*)alloc((size_t)ROWS_ * E_ * 2);
  _Float16* kc[6]; _Float16* vc[6];
  for (int l = 0; l < 6; ++l) {
    kc[l] = (_Float16*)alloc((size_t)ROWS_ * E_ * 2);
    vc[l] = (_Float16*)alloc((size_t)ROWS_ * E_ * 2);
  }
  int* pred = (int*)alloc((size_t)T_ * B_ * 4);

  float* tok_out = (float*)d_out;              // [T,B] tokens (as float)
  float* dists   = tok_out + T_ * B_;          // [T,B,V]

  const size_t ATTN_LDS = 18560;
  auto gemm = [&](const _Float16* A, const _Float16* W, const float* bias,
                  float* o32, _Float16* o16, int M, int N, int K, int relu) {
    if (M >= 32) {
      dim3 g((N + 63) / 64, (M + 31) / 32);
      k_gemm<2><<<g, 128, 0, stream>>>(A, W, bias, o32, o16, M, N, K, relu);
    } else {
      dim3 g((N + 63) / 64, (M + 15) / 16);
      k_gemm<1><<<g, 128, 0, stream>>>(A, W, bias, o32, o16, M, N, K, relu);
    }
  };
  auto resln = [&](const float* y, const _Float16* xr, const float* w,
                   const float* b, _Float16* o) {
    k_res_ln<<<ROWS_, 256, 0, stream>>>(y, xr, w, b, o);
  };

  k_init<<<1024, 256, 0, stream>>>(pred, tok_out, dists);

  // ---------------- encoder ----------------
  k_embed_enc<<<S_ * B_, 256, 0, stream>>>(input_tokens, enc_emb, xe);
  for (int l = 0; l < 6; ++l) {
    const EL& p = enc[l]; const EL16& w = enc16[l];
    gemm(xe, w.sa.wq, p.sa.bq, nullptr, qb, ROWS_, E_, E_, 0);
    gemm(xe, w.sa.wk, p.sa.bk, nullptr, kb, ROWS_, E_, E_, 0);
    gemm(xe, w.sa.wv, p.sa.bv, nullptr, vb, ROWS_, E_, E_, 0);
    k_attn<<<NH_ * B_, 64, ATTN_LDS, stream>>>(qb, kb, vb, at, 0);
    gemm(at, w.sa.wo, p.sa.bo, y32, nullptr, ROWS_, E_, E_, 0);
    resln(y32, xe, p.ln1w, p.ln1b, xe);
    gemm(xe, w.w1, p.b1, nullptr, hb, ROWS_, FF_, E_, 1);
    gemm(hb, w.w2, p.b2, y32, nullptr, ROWS_, E_, FF_, 0);
    resln(y32, xe, p.ln2w, p.ln2b, xe);
  }
  resln(nullptr, xe, enc_nw, enc_nb, mem);

  // cross-attention K/V are loop-invariant: compute once per decoder layer
  for (int l = 0; l < 6; ++l) {
    gemm(mem, dec16[l].ca.wk, dec[l].ca.bk, nullptr, kc[l], ROWS_, E_, E_, 0);
    gemm(mem, dec16[l].ca.wv, dec[l].ca.bv, nullptr, vc[l], ROWS_, E_, E_, 0);
  }

  // ---------------- greedy decode ----------------
  for (int i = 0; i < T_ - 1; ++i) {
    k_embed_dec<<<T_ * B_, 256, 0, stream>>>(pred, dec_emb, xd);
    for (int l = 0; l < 6; ++l) {
      const DL& p = dec[l]; const DL16& w = dec16[l];
      // causal self-attention
      gemm(xd, w.sa.wq, p.sa.bq, nullptr, qb, ROWS_, E_, E_, 0);
      gemm(xd, w.sa.wk, p.sa.bk, nullptr, kb, ROWS_, E_, E_, 0);
      gemm(xd, w.sa.wv, p.sa.bv, nullptr, vb, ROWS_, E_, E_, 0);
      k_attn<<<NH_ * B_, 64, ATTN_LDS, stream>>>(qb, kb, vb, at, 1);
      gemm(at, w.sa.wo, p.sa.bo, y32, nullptr, ROWS_, E_, E_, 0);
      resln(y32, xd, p.ln1w, p.ln1b, xd);
      // cross attention (K/V precomputed)
      gemm(xd, w.ca.wq, p.ca.bq, nullptr, qb, ROWS_, E_, E_, 0);
      k_attn<<<NH_ * B_, 64, ATTN_LDS, stream>>>(qb, kc[l], vc[l], at, 0);
      gemm(at, w.ca.wo, p.ca.bo, y32, nullptr, ROWS_, E_, E_, 0);
      resln(y32, xd, p.ln2w, p.ln2b, xd);
      // FFN
      gemm(xd, w.w1, p.b1, nullptr, hb, ROWS_, FF_, E_, 1);
      gemm(hb, w.w2, p.b2, y32, nullptr, ROWS_, E_, FF_, 0);
      resln(y32, xd, p.ln3w, p.ln3b, xd);
    }
    resln(nullptr, xd, dec_nw, dec_nb, xn);
    // logits for row i only: M=8 valid rows in one 16-row WMMA tile
    float* drow = dists + (size_t)(i + 1) * B_ * V_;
    gemm(xn + (size_t)i * B_ * E_, proj16, proj_b, drow, nullptr, B_, V_, E_, 0);
    k_argmax<<<B_, 256, 0, stream>>>(drow, pred, tok_out, i);
  }
}
